// ReNetLayer_3582002725266
// MI455X (gfx1250) — matrix operand order repaired and checked by reading, not compiled, not run on hardware
//
#include <hip/hip_runtime.h>
#include <hip/hip_bf16.h>

typedef __attribute__((ext_vector_type(16))) _Float16 v16h;
typedef __attribute__((ext_vector_type(8)))  _Float16 v8h;
typedef __attribute__((ext_vector_type(8)))  float    v8f;
typedef __attribute__((ext_vector_type(4)))  float    v4f;

#define HID    64
#define G4     256          // 4*HID gate columns
#define T128   128          // J == I == 128 (sequence length for both sweeps)
#define NSEQ   2048         // B*I == B*J
#define DV     12           // vertical input dim
#define DH     128          // horizontal input dim (2*HID)
#define XPADV  16           // vertical input padded to 16 halves
#define HSTR_V 104          // vertical hbuf row stride: 64 h | 32 x(12+pad0) | 8 pad (208B, 16B aligned)
#define HSTR_H 200          // horizontal hbuf row stride: 64 h | 128 x | 8 pad (400B, 16B aligned)
#define ZSTR   20           // zbuf column stride in floats (80B, 16B aligned)

// ---- branch-free transcendentals on the native TRANS32 pipe
__device__ __forceinline__ float fast_rcp(float x) { return __builtin_amdgcn_rcpf(x); }

__device__ __forceinline__ float sigmoidf_(float x) {
    return fast_rcp(1.0f + __expf(-x));
}

#if __has_builtin(__builtin_amdgcn_tanhf)
__device__ __forceinline__ float fast_tanh(float x) { return __builtin_amdgcn_tanhf(x); }
#else
__device__ __forceinline__ float fast_tanh(float x) {
    // 1 - 2/(e^{2x}+1): monotone, exact limits at +/-inf, no branches
    return 1.0f - 2.0f * fast_rcp(__expf(2.0f * x) + 1.0f);
}
#endif

// ---- A-operand fragment from a row-major [16][stride] f16 LDS tile.
// ISA layout (16-bit A 16x32): lanes 0-15: M=lane, K=k0+0..7 (v0..3), k0+16..23 (v4..7)
//                              lanes 16-31: M=lane-16, K=k0+8..15, k0+24..31
__device__ __forceinline__ v16h ldsA(const _Float16* tile, int stride_h, int k0) {
    int lane = threadIdx.x & 31;
    int m    = lane & 15;
    int cb   = k0 + ((lane >> 4) << 3);
    const _Float16* rp = tile + m * stride_h;
    v8h lo = *(const v8h*)(rp + cb);
    v8h hi = *(const v8h*)(rp + cb + 16);
    v16h a;
#pragma unroll
    for (int i = 0; i < 8; ++i) { a[i] = lo[i]; a[i + 8] = hi[i]; }
    return a;
}

// ---- B-operand fragment: prebuilt per-lane layout, 16 halves (32B) per lane.
__device__ __forceinline__ v16h ldsB(const _Float16* ufrag, int kc, int nt) {
    int lane = threadIdx.x & 31;
    const _Float16* p = ufrag + ((((kc * 16 + nt) * 32) + lane) << 4);
    v8h lo = *(const v8h*)p;
    v8h hi = *(const v8h*)(p + 8);
    v16h b;
#pragma unroll
    for (int i = 0; i < 8; ++i) { b[i] = lo[i]; b[i + 8] = hi[i]; }
    return b;
}

// Build B fragments for Ucomb = [U (HID rows); W (D rows, zero padded)] of shape [KC*32, 256].
// ISA B layout (16-bit 32x16): lanes 0-15: N=lane, K=0..7 (v0..3), 16..23 (v4..7); lanes 16-31: K=8..15, 24..31.
__device__ void build_ufrag(_Float16* ufrag, const float* __restrict__ U,
                            const float* __restrict__ W, int D, int KC, int tid) {
    int total = KC * 16 * 32 * 8;   // dwords
    for (int e = tid; e < total; e += 256) {
        int dw   = e & 7;
        int lane = (e >> 3) & 31;
        int nt   = (e >> 8) & 15;
        int kc   = e >> 12;
        int klo  = ((lane & 16) ? 8 : 0) + ((dw & 4) ? 16 : 0) + ((dw & 3) << 1);
        int k    = kc * 32 + klo;
        int n    = (nt << 4) + (lane & 15);
        float f0 = 0.f, f1 = 0.f;
        if (k < HID) {
            f0 = U[k * G4 + n];
            f1 = U[(k + 1) * G4 + n];
        } else {
            int kr = k - HID;
            if (kr < D)     f0 = W[kr * G4 + n];
            if (kr + 1 < D) f1 = W[(kr + 1) * G4 + n];
        }
        int idx = ((((kc * 16 + nt) * 32) + lane) << 4) + (dw << 1);
        ufrag[idx]     = (_Float16)f0;
        ufrag[idx + 1] = (_Float16)f1;
    }
}

// ---- Pack inputs [B,256,256,3] f32 -> xv [NSEQ][128][16] f16 (12 real + 4 zero pad)
__global__ __launch_bounds__(256) void pack_x_kernel(const float* __restrict__ in,
                                                     _Float16* __restrict__ xv) {
    int idx = blockIdx.x * 256 + threadIdx.x;
    if (idx >= NSEQ * T128 * XPADV) return;
    int d = idx & 15;
    int t = (idx >> 4) & 127;
    int n = idx >> 11;
    int b = n >> 7, ii = n & 127;
    float v = 0.f;
    if (d < DV) {
        int pr = d / 6, rem = d % 6, pc = rem / 3, c = rem % 3;
        v = in[((((size_t)b * 256) + (t * 2 + pr)) * 256 + (ii * 2 + pc)) * 3 + c];
    }
    xv[idx] = (_Float16)v;
}

__device__ __forceinline__ void store_ztile(float* zbuf, int nt, const v8f& acc) {
    int lane = threadIdx.x & 31;
    int sel  = (lane >> 4) & 1;
    int col  = nt * 16 + (lane & 15);
    float* zp = zbuf + col * ZSTR + sel * 8;
    v4f lo = {acc[0], acc[1], acc[2], acc[3]};
    v4f hi = {acc[4], acc[5], acc[6], acc[7]};
    *(v4f*)zp       = lo;
    *(v4f*)(zp + 4) = hi;
}

// ---- Vertical sweep: both directions (blockIdx.x>>7 selects), 16 sequences per WG.
__global__ __launch_bounds__(256) void renet_lstm_vert(
    const _Float16* __restrict__ xv,
    const float* __restrict__ Wf, const float* __restrict__ Uf, const float* __restrict__ bf,
    const float* __restrict__ Wr, const float* __restrict__ Ur, const float* __restrict__ br,
    _Float16* __restrict__ vbuf) {
    extern __shared__ char smem[];
    _Float16* ufrag = (_Float16*)smem;                       // 3*16*32*16 halves = 49152 B
    _Float16* xpad  = (_Float16*)(smem + 49152);             // 128*16*16 halves  = 65536 B
    _Float16* hbuf  = (_Float16*)(smem + 49152 + 65536);     // 16*HSTR_V halves  = 3328 B
    float*    zbuf  = (float*)(smem + 49152 + 65536 + 3328); // 256*ZSTR f32      = 20480 B

    int tid  = threadIdx.x;
    int dir  = blockIdx.x >> 7;
    int wg   = blockIdx.x & 127;
    const float* W  = dir ? Wr : Wf;
    const float* U  = dir ? Ur : Uf;
    const float* bb = dir ? br : bf;
    int dofs = dir ? HID : 0;
    int nbase = wg << 4;

    build_ufrag(ufrag, U, W, DV, 3, tid);

    // preload this block's entire input sequence: [t][r][16] halves
    for (int e = tid; e < T128 * 16 * 2; e += 256) {
        int part = e & 1;
        int r    = (e >> 1) & 15;
        int t    = e >> 5;
        const v8h* src = (const v8h*)(xv + (((size_t)(nbase + r)) * T128 + t) * XPADV + part * 8);
        *(v8h*)(xpad + (t * 16 + r) * XPADV + part * 8) = *src;
    }
    // zero whole hbuf (h zone, x zone incl. pad-zeros at cols 76..95)
    for (int e = tid; e < 16 * HSTR_V; e += 256) hbuf[e] = (_Float16)0.f;

    __syncthreads();

    int lane = tid & 31;
    int wave = tid >> 5;

    // Hoist loop-invariant B fragments + bias into registers (ufrag LDS dead after this).
    v16h Bf[2][3];
    float biasv[2];
#pragma unroll
    for (int p = 0; p < 2; ++p) {
        int nt = wave * 2 + p;
        biasv[p] = bb[nt * 16 + (lane & 15)];
#pragma unroll
        for (int kc = 0; kc < 3; ++kc) Bf[p][kc] = ldsB(ufrag, kc, nt);
    }
    // copy x for t=0 into hbuf x-zone (cols 64..75)
    if (tid < 16 * DV) {
        int r = tid / DV, d = tid % DV;
        int tin0 = dir ? (T128 - 1) : 0;
        hbuf[r * HSTR_V + 64 + d] = xpad[(tin0 * 16 + r) * XPADV + d];
    }

    __syncthreads();

    int j  = tid & 63;
    int r0 = tid >> 6;
    float cst[4] = {0.f, 0.f, 0.f, 0.f};

    for (int t = 0; t < T128; ++t) {
        // per-step combined operand [h | x] -> 3 uniform A fragments, loaded once
        v16h afr[3];
#pragma unroll
        for (int kc = 0; kc < 3; ++kc) afr[kc] = ldsA(hbuf, HSTR_V, kc * 32);
#pragma unroll
        for (int p = 0; p < 2; ++p) {
            v8f acc;
#pragma unroll
            for (int v = 0; v < 8; ++v) acc[v] = biasv[p];
#pragma unroll
            for (int kc = 0; kc < 3; ++kc)
                acc = __builtin_amdgcn_wmma_f32_16x16x32_f16(false, afr[kc], false, Bf[p][kc],
                                                             (short)0, acc, false, false);
            store_ztile(zbuf, wave * 2 + p, acc);
        }
        __syncthreads();
#pragma unroll
        for (int q = 0; q < 4; ++q) {
            int r = r0 + q * 4;
            float zi = zbuf[j * ZSTR + r];
            float zf = zbuf[(j + 64) * ZSTR + r];
            float zg = zbuf[(j + 128) * ZSTR + r];
            float zo = zbuf[(j + 192) * ZSTR + r];
            float ig = sigmoidf_(zi);
            float fg = sigmoidf_(zf);
            float gg = fast_tanh(zg);
            float og = sigmoidf_(zo);
            float c  = fg * cst[q] + ig * gg;
            cst[q]   = c;
            float h  = og * fast_tanh(c);
            hbuf[r * HSTR_V + j] = (_Float16)h;
            int n = nbase + r;
            int b = n >> 7, col = n & 127;
            vbuf[((((size_t)b * T128 + t) * T128 + col) << 7) + dofs + j] = (_Float16)h;
        }
        // copy x for t+1 into hbuf x-zone (its old contents were read before last barrier)
        if (t + 1 < T128 && tid < 16 * DV) {
            int r = tid / DV, d = tid % DV;
            int tin = dir ? (T128 - 2 - t) : (t + 1);
            hbuf[r * HSTR_V + 64 + d] = xpad[(tin * 16 + r) * XPADV + d];
        }
        __syncthreads();
    }
}

// ---- Horizontal sweep: input = vbuf [B,J,I,128] f16, output = out [B,J,I,128] f32.
__global__ __launch_bounds__(256) void renet_lstm_horz(
    const _Float16* __restrict__ vbuf,
    const float* __restrict__ Wf, const float* __restrict__ Uf, const float* __restrict__ bf,
    const float* __restrict__ Wr, const float* __restrict__ Ur, const float* __restrict__ br,
    float* __restrict__ out) {
    extern __shared__ char smem[];
    _Float16* ufrag = (_Float16*)smem;                    // 6*16*32*16 halves = 98304 B
    _Float16* hbuf  = (_Float16*)(smem + 98304);          // 16*HSTR_H halves  = 6400 B
    float*    zbuf  = (float*)(smem + 98304 + 6400);      // 20480 B

    int tid  = threadIdx.x;
    int dir  = blockIdx.x >> 7;
    int wg   = blockIdx.x & 127;
    const float* W  = dir ? Wr : Wf;
    const float* U  = dir ? Ur : Uf;
    const float* bb = dir ? br : bf;
    int dofs = dir ? HID : 0;
    int nbase = wg << 4;

    build_ufrag(ufrag, U, W, DH, 6, tid);
    for (int e = tid; e < 16 * HSTR_H; e += 256) hbuf[e] = (_Float16)0.f;

    // stage x for t = 0 directly into hbuf x-zone (cols 64..191)
    {
        int r = tid >> 4, part = tid & 15;
        int n2 = nbase + r, b = n2 >> 7, jj = n2 & 127;
        int tin0 = dir ? (T128 - 1) : 0;
        const v8h* s = (const v8h*)(vbuf + (((size_t)(b * T128 + jj)) * T128 + tin0) * DH + part * 8);
        *(v8h*)(hbuf + r * HSTR_H + 64 + part * 8) = *s;
    }

    __syncthreads();

    int lane = tid & 31;
    int wave = tid >> 5;

    // Hoist loop-invariant B fragments + bias into registers.
    v16h Bf[2][6];
    float biasv[2];
#pragma unroll
    for (int p = 0; p < 2; ++p) {
        int nt = wave * 2 + p;
        biasv[p] = bb[nt * 16 + (lane & 15)];
#pragma unroll
        for (int kc = 0; kc < 6; ++kc) Bf[p][kc] = ldsB(ufrag, kc, nt);
    }

    int j  = tid & 63;
    int r0 = tid >> 6;
    float cst[4] = {0.f, 0.f, 0.f, 0.f};

    for (int t = 0; t < T128; ++t) {
        // combined [h | x] operand -> 6 uniform A fragments, loaded once per step
        v16h afr[6];
#pragma unroll
        for (int kc = 0; kc < 6; ++kc) afr[kc] = ldsA(hbuf, HSTR_H, kc * 32);
#pragma unroll
        for (int p = 0; p < 2; ++p) {
            v8f acc;
#pragma unroll
            for (int v = 0; v < 8; ++v) acc[v] = biasv[p];
#pragma unroll
            for (int kc = 0; kc < 6; ++kc)
                acc = __builtin_amdgcn_wmma_f32_16x16x32_f16(false, afr[kc], false, Bf[p][kc],
                                                             (short)0, acc, false, false);
            store_ztile(zbuf, wave * 2 + p, acc);
        }
        __syncthreads();
        // stage next step's x into hbuf x-zone (overlaps global load with gate phase;
        // previous contents were last read before the barrier above)
        if (t + 1 < T128) {
            int r = tid >> 4, part = tid & 15;
            int n2 = nbase + r, b = n2 >> 7, jj = n2 & 127;
            int tin = dir ? (T128 - 2 - t) : (t + 1);
            const v8h* s = (const v8h*)(vbuf + (((size_t)(b * T128 + jj)) * T128 + tin) * DH + part * 8);
            *(v8h*)(hbuf + r * HSTR_H + 64 + part * 8) = *s;
        }
#pragma unroll
        for (int q = 0; q < 4; ++q) {
            int r = r0 + q * 4;
            float zi = zbuf[j * ZSTR + r];
            float zf = zbuf[(j + 64) * ZSTR + r];
            float zg = zbuf[(j + 128) * ZSTR + r];
            float zo = zbuf[(j + 192) * ZSTR + r];
            float ig = sigmoidf_(zi);
            float fg = sigmoidf_(zf);
            float gg = fast_tanh(zg);
            float og = sigmoidf_(zo);
            float c  = fg * cst[q] + ig * gg;
            cst[q]   = c;
            float h  = og * fast_tanh(c);
            hbuf[r * HSTR_H + j] = (_Float16)h;
            int n = nbase + r;
            int b = n >> 7, jj = n & 127;
            out[(((size_t)(b * T128 + jj)) * T128 + t) * DH + dofs + j] = h;
        }
        __syncthreads();
    }
}

extern "C" void kernel_launch(void* const* d_in, const int* in_sizes, int n_in,
                              void* d_out, int out_size, void* d_ws, size_t ws_size,
                              hipStream_t stream) {
    const float* inputs = (const float*)d_in[0];
    const float* W_ud = (const float*)d_in[1];
    const float* U_ud = (const float*)d_in[2];
    const float* b_ud = (const float*)d_in[3];
    const float* W_du = (const float*)d_in[4];
    const float* U_du = (const float*)d_in[5];
    const float* b_du = (const float*)d_in[6];
    const float* W_lr = (const float*)d_in[7];
    const float* U_lr = (const float*)d_in[8];
    const float* b_lr = (const float*)d_in[9];
    const float* W_rl = (const float*)d_in[10];
    const float* U_rl = (const float*)d_in[11];
    const float* b_rl = (const float*)d_in[12];

    _Float16* xv   = (_Float16*)d_ws;                                   // 8 MB
    size_t xv_bytes = (size_t)NSEQ * T128 * XPADV * sizeof(_Float16);
    _Float16* vbuf = (_Float16*)((char*)d_ws + ((xv_bytes + 255) & ~(size_t)255)); // 64 MB

    // 1) pack inputs into padded f16 vertical sequences
    int npack = NSEQ * T128 * XPADV;
    pack_x_kernel<<<(npack + 255) / 256, 256, 0, stream>>>(inputs, xv);

    // 2) vertical bidirectional LSTM (dir encoded in blockIdx.x>>7)
    size_t smem_v = 49152 + 65536 + 3328 + 20480;
    renet_lstm_vert<<<256, 256, smem_v, stream>>>(xv, W_ud, U_ud, b_ud,
                                                  W_du, U_du, b_du, vbuf);

    // 3) horizontal bidirectional LSTM -> final f32 output
    size_t smem_h = 98304 + 6400 + 20480;
    renet_lstm_horz<<<256, 256, smem_h, stream>>>(vbuf, W_lr, U_lr, b_lr,
                                                  W_rl, U_rl, b_rl, (float*)d_out);
}
